// BUrumorGAT_25494925869741
// MI455X (gfx1250) — matrix operand order repaired and checked by reading, not compiled, not run on hardware
//
#include <hip/hip_runtime.h>
#include <cstdint>
#include <cstddef>

// ---------------- problem constants (match reference) ----------------
constexpr int kN    = 50000;
constexpr int kE    = 800000;
constexpr int kB    = 512;
constexpr int kIN   = 768;
constexpr int kHID  = 256;
constexpr int kOUT  = 128;
constexpr int kH    = 4;
constexpr int kD1   = kHID / kH;   // 64
constexpr int kD2   = kOUT / kH;   // 32
constexpr int kXC   = kHID + kIN;  // 1024
constexpr int kFEAT = kOUT + kHID; // 384
constexpr int kEtot = kE + kN;     // 850000 (self loops appended)

typedef __attribute__((ext_vector_type(16))) __bf16 v16bf;
typedef __attribute__((ext_vector_type(8)))  float  v8f;

struct __align__(16) U4 { unsigned int a, b, c, d; };
union FragAB { v16bf v; unsigned short u[16]; U4 q[2]; };

// ---- CDNA5 async global->LDS staging (fallback to sync staging if absent) ----
#if defined(__gfx1250__) && __has_builtin(__builtin_amdgcn_global_load_async_to_lds_b128)
#define USE_ASYNC_LDS 1
#else
#define USE_ASYNC_LDS 0
#endif

// builtin signature (from clang diagnostic): arg0 = v4i __device__(AS1)*, then LDS dst, imm, imm
typedef int v4i __attribute__((vector_size(16)));
typedef __attribute__((address_space(1))) v4i* gbl_v4i_ptr;
typedef __attribute__((address_space(3))) v4i* lds_v4i_ptr;

__device__ __forceinline__ gbl_v4i_ptr gbl_cast(const unsigned short* p) {
  // global generic address == AS1 address
  return (gbl_v4i_ptr)(unsigned long long)p;
}
__device__ __forceinline__ lds_v4i_ptr lds_cast(unsigned short* p) {
  // generic LDS address = {shared aperture, lds_offset[31:0]} -> AS3 is low 32 bits
  return (lds_v4i_ptr)(unsigned int)(unsigned long long)p;
}

__device__ __forceinline__ void wait_async_zero() {
#if defined(__gfx1250__) && __has_builtin(__builtin_amdgcn_s_wait_asynccnt)
  __builtin_amdgcn_s_wait_asynccnt(0);
#else
  asm volatile("s_wait_asynccnt 0" ::: "memory");
#endif
}

__device__ __forceinline__ unsigned short f2bf(float f) {
  unsigned int u = __float_as_uint(f);
  u += 0x7fffu + ((u >> 16) & 1u);      // round-to-nearest-even
  return (unsigned short)(u >> 16);
}

// monotonic-int trick: works for mixed signs with -inf init
__device__ __forceinline__ void atomicMaxFloat(float* addr, float val) {
  if (val >= 0.f) atomicMax((int*)addr, __float_as_int(val));
  else            atomicMin((unsigned int*)addr, __float_as_uint(val));
}

// ---------------- elementwise helpers ----------------
__global__ void k_cvt_bf16(const float* __restrict__ src,
                           unsigned short* __restrict__ dst, int n) {
  int i = blockIdx.x * blockDim.x + threadIdx.x;
  if (i < n) dst[i] = f2bf(src[i]);
}

// convert + transpose: src is [K][N] row-major fp32, dst is [N][K] row-major bf16
__global__ void k_cvt_bf16_t(const float* __restrict__ src,
                             unsigned short* __restrict__ dst, int K, int N) {
  int i = blockIdx.x * blockDim.x + threadIdx.x;
  if (i >= K * N) return;
  int k = i / N, n = i - k * N;
  dst[(size_t)n * K + k] = f2bf(src[i]);
}

__global__ void k_init_f32(float* __restrict__ p, float v, int n) {
  int i = blockIdx.x * blockDim.x + threadIdx.x;
  if (i < n) p[i] = v;
}

__global__ void k_init_i32(int* __restrict__ p, int v, int n) {
  int i = blockIdx.x * blockDim.x + threadIdx.x;
  if (i < n) p[i] = v;
}

// ------- bf16 WMMA GEMM: C[M,Nc] = A[M,K] * Bt[Nc,K]^T  (Bt = B transposed) -------
// block tile 128x128, 8 waves (4 m x 2 n), wave tile 32x64 = eight 16x16x32 WMMAs
#define GBM 128
#define GBN 128
#define GBK 32
#define TSTR 40   // GBK + 8 ushorts pad (row stride 80 B, 16B aligned, conflict-free)

__global__ __launch_bounds__(256)
void k_gemm_bf16(const unsigned short* __restrict__ A,
                 const unsigned short* __restrict__ Bt,
                 float* __restrict__ C, int M, int Nc, int K) {
  __shared__ __align__(16) unsigned short As[GBM * TSTR]; // 10240 B
  __shared__ __align__(16) unsigned short Bs[GBN * TSTR]; // 10240 B

  const int tid  = threadIdx.x;
  const int wave = tid >> 5;
  const int lane = tid & 31;
  const int wm   = wave >> 1;     // 0..3 -> 32 rows each
  const int wn   = wave & 1;      // 0..1 -> 64 cols each
  const int half = lane >> 4;
  const int mrow = lane & 15;
  const int m0   = blockIdx.x * GBM;
  const int n0   = blockIdx.y * GBN;

  const v8f vzero = {0.f,0.f,0.f,0.f,0.f,0.f,0.f,0.f};
  v8f acc[2][4];
  #pragma unroll
  for (int mt = 0; mt < 2; ++mt)
    #pragma unroll
    for (int nt = 0; nt < 4; ++nt) acc[mt][nt] = vzero;

  for (int k0 = 0; k0 < K; k0 += GBK) {
    __syncthreads();   // previous iteration's fragment reads complete
#if USE_ASYNC_LDS
    // direct global->LDS DMA (ASYNCcnt); no VGPR round-trip.
    // OOB A rows are skipped: stale LDS only feeds output rows m>=M, never stored.
    #pragma unroll
    for (int c = 0; c < 2; ++c) {
      int ch  = tid + c * 256;
      int row = ch >> 2;            // 4 chunks of 8 bf16 per row
      int kc  = (ch & 3) << 3;
      int gm  = m0 + row;
      if (gm < M)
        __builtin_amdgcn_global_load_async_to_lds_b128(
            gbl_cast(A + (size_t)gm * K + k0 + kc),
            lds_cast(&As[row * TSTR + kc]), 0, 0);
      __builtin_amdgcn_global_load_async_to_lds_b128(
          gbl_cast(Bt + (size_t)(n0 + row) * K + k0 + kc),
          lds_cast(&Bs[row * TSTR + kc]), 0, 0);
    }
    wait_async_zero();   // this wave's DMAs landed
    __syncthreads();     // all waves' DMAs landed
#else
    // stage A tile: 128x32 bf16 = 512 16B-chunks, 2 per thread (guard M edge)
    #pragma unroll
    for (int c = 0; c < 2; ++c) {
      int ch  = tid + c * 256;
      int row = ch >> 2;
      int kc  = (ch & 3) << 3;
      U4 val = {0u, 0u, 0u, 0u};
      int gm = m0 + row;
      if (gm < M) val = *(const U4*)(A + (size_t)gm * K + k0 + kc);
      *(U4*)(&As[row * TSTR + kc]) = val;
    }
    #pragma unroll
    for (int c = 0; c < 2; ++c) {
      int ch  = tid + c * 256;
      int row = ch >> 2;
      int kc  = (ch & 3) << 3;
      U4 val = *(const U4*)(Bt + (size_t)(n0 + row) * K + k0 + kc);
      *(U4*)(&Bs[row * TSTR + kc]) = val;
    }
    __syncthreads();
#endif

    // A frags: lane m = mrow, elem i -> k = (i/8)*16 + half*8 + i%8
    FragAB af[2];
    #pragma unroll
    for (int mt = 0; mt < 2; ++mt) {
      int row = wm * 32 + mt * 16 + mrow;
      af[mt].q[0] = *(const U4*)(&As[row * TSTR + half * 8]);
      af[mt].q[1] = *(const U4*)(&As[row * TSTR + 16 + half * 8]);
    }
    // B frags: lane n = mrow, elem i -> k = (i/8)*16 + half*8 + i%8
    FragAB bfr[4];
    #pragma unroll
    for (int nt = 0; nt < 4; ++nt) {
      int nrow = wn * 64 + nt * 16 + mrow;
      bfr[nt].q[0] = *(const U4*)(&Bs[nrow * TSTR + half * 8]);
      bfr[nt].q[1] = *(const U4*)(&Bs[nrow * TSTR + 16 + half * 8]);
    }
    #pragma unroll
    for (int mt = 0; mt < 2; ++mt)
      #pragma unroll
      for (int nt = 0; nt < 4; ++nt)
        acc[mt][nt] = __builtin_amdgcn_wmma_f32_16x16x32_bf16(
            false, af[mt].v, false, bfr[nt].v, (short)0, acc[mt][nt], false, false);
  }

  // epilogue: C/D layout lane n = lane%16, vgpr r -> m = r + 8*half
  #pragma unroll
  for (int mt = 0; mt < 2; ++mt)
    #pragma unroll
    for (int nt = 0; nt < 4; ++nt) {
      int n = n0 + wn * 64 + nt * 16 + mrow;
      #pragma unroll
      for (int r = 0; r < 8; ++r) {
        int m = m0 + wm * 32 + mt * 16 + r + half * 8;
        if (m < M) C[(size_t)m * Nc + n] = acc[mt][nt][r];
      }
    }
}

// ---------------- attention coefficients a_s/a_d per (node, head) ----------------
__global__ void k_att(const float* __restrict__ hpre, const float* __restrict__ asv,
                      const float* __restrict__ adv, float* __restrict__ oas,
                      float* __restrict__ oad, int hdim) {
  int idx = blockIdx.x * blockDim.x + threadIdx.x;
  if (idx >= kN * kH) return;
  int n = idx >> 2, h = idx & 3;
  const float* hp = hpre + (size_t)n * (kH * hdim) + h * hdim;
  float s = 0.f, t = 0.f;
  for (int k = 0; k < hdim; ++k) {
    float hv = hp[k];
    s += hv * asv[h * hdim + k];
    t += hv * adv[h * hdim + k];
  }
  oas[idx] = s; oad[idx] = t;
}

// ---------------- edge softmax passes ----------------
__device__ __forceinline__ void edge_sd(int e, const int* __restrict__ ei, int& s, int& dn) {
  if (e < kE) { s = ei[kE + e]; dn = ei[e]; }   // src = row1, dst = row0 (BU swap)
  else        { s = dn = e - kE; }              // self loops
}

__global__ void k_edge_max(const int* __restrict__ ei, const float* __restrict__ as_,
                           const float* __restrict__ ad_, float* __restrict__ emax) {
  int idx = blockIdx.x * blockDim.x + threadIdx.x;
  if (idx >= kEtot * kH) return;
  int e = idx >> 2, h = idx & 3;
  int s, dn; edge_sd(e, ei, s, dn);
  float v = as_[s * kH + h] + ad_[dn * kH + h];
  v = v > 0.f ? v : 0.2f * v;                   // LeakyReLU(0.2)
  atomicMaxFloat(&emax[dn * kH + h], v);
}

__global__ void k_edge_exp(const int* __restrict__ ei, const float* __restrict__ as_,
                           const float* __restrict__ ad_, const float* __restrict__ emax,
                           float* __restrict__ denom, float* __restrict__ expe) {
  int idx = blockIdx.x * blockDim.x + threadIdx.x;
  if (idx >= kEtot * kH) return;
  int e = idx >> 2, h = idx & 3;
  int s, dn; edge_sd(e, ei, s, dn);
  float v = as_[s * kH + h] + ad_[dn * kH + h];
  v = v > 0.f ? v : 0.2f * v;
  float ex = __expf(v - emax[dn * kH + h]);
  expe[(size_t)e * kH + h] = ex;
  atomicAdd(&denom[dn * kH + h], ex);
}

// wave per edge; lane strides channels; scatter-add into L2
__global__ __launch_bounds__(256)
void k_edge_agg(const int* __restrict__ ei, const float* __restrict__ hpre,
                const float* __restrict__ expe, const float* __restrict__ denom,
                float* __restrict__ agg, int hid, int hdim) {
  int e = blockIdx.x * 8 + (threadIdx.x >> 5);
  if (e >= kEtot) return;
  int lane = threadIdx.x & 31;
  int s, dn; edge_sd(e, ei, s, dn);
  for (int c = lane; c < hid; c += 32) {
    int h = c / hdim;
    float alpha = expe[(size_t)e * kH + h] / (denom[(size_t)dn * kH + h] + 1e-16f);
    atomicAdd(&agg[(size_t)dn * hid + c], hpre[(size_t)s * hid + c] * alpha);
  }
}

__global__ void k_bias(float* __restrict__ h, const float* __restrict__ b,
                       int cols, int relu) {
  int idx = blockIdx.x * blockDim.x + threadIdx.x;
  if (idx >= kN * cols) return;
  float v = h[idx] + b[idx % cols];
  if (relu) v = fmaxf(v, 0.f);
  h[idx] = v;
}

__global__ void k_roots(const int* __restrict__ batch, int* __restrict__ roots) {
  int n = blockIdx.x * blockDim.x + threadIdx.x;
  if (n < kN) atomicMin(&roots[batch[n]], n);
}

// xc = relu(concat(h1, x[root])) packed to bf16 for GEMM2
__global__ void k_build_xc(const float* __restrict__ h1, const float* __restrict__ x,
                           const int* __restrict__ batch, const int* __restrict__ roots,
                           unsigned short* __restrict__ xcb) {
  int idx = blockIdx.x * blockDim.x + threadIdx.x;
  if (idx >= kN * kXC) return;
  int n = idx >> 10, c = idx & 1023;
  float v;
  if (c < kHID) v = h1[(size_t)n * kHID + c];
  else          v = x[(size_t)roots[batch[n]] * kIN + (c - kHID)];
  xcb[idx] = f2bf(fmaxf(v, 0.f));
}

__global__ void k_pool(const float* __restrict__ h2, const float* __restrict__ h1,
                       const int* __restrict__ batch, const int* __restrict__ roots,
                       float* __restrict__ out, float* __restrict__ counts) {
  int idx = blockIdx.x * blockDim.x + threadIdx.x;
  if (idx >= kN * kFEAT) return;
  int n = idx / kFEAT, c = idx - n * kFEAT;
  int b = batch[n];
  float v = (c < kOUT) ? h2[(size_t)n * kOUT + c]
                       : h1[(size_t)roots[b] * kHID + (c - kOUT)];
  atomicAdd(&out[b * kFEAT + c], v);
  if (c == 0) atomicAdd(&counts[b], 1.0f);
}

__global__ void k_pool_div(float* __restrict__ out, const float* __restrict__ counts) {
  int idx = blockIdx.x * blockDim.x + threadIdx.x;
  if (idx >= kB * kFEAT) return;
  out[idx] /= fmaxf(counts[idx / kFEAT], 1.0f);
}

// ---------------- launch ----------------
static inline int cdiv(int a, int b) { return (a + b - 1) / b; }

extern "C" void kernel_launch(void* const* d_in, const int* in_sizes, int n_in,
                              void* d_out, int out_size, void* d_ws, size_t ws_size,
                              hipStream_t stream) {
  (void)in_sizes; (void)n_in; (void)out_size; (void)ws_size;
  const float* x      = (const float*)d_in[0];
  const int*   ei     = (const int*)d_in[1];
  const int*   batch  = (const int*)d_in[2];
  const float* W1     = (const float*)d_in[3];
  const float* att_s1 = (const float*)d_in[4];
  const float* att_d1 = (const float*)d_in[5];
  const float* b1     = (const float*)d_in[6];
  const float* W2     = (const float*)d_in[7];
  const float* att_s2 = (const float*)d_in[8];
  const float* att_d2 = (const float*)d_in[9];
  const float* b2     = (const float*)d_in[10];
  float* out = (float*)d_out;

  // workspace arena (~248 MB), regions reused across phases
  char* w = (char*)d_ws;
  auto alloc = [&](size_t bytes) -> void* {
    void* p = (void*)w; w += (bytes + 255) & ~(size_t)255; return p;
  };
  unsigned short* xb    = (unsigned short*)alloc((size_t)kN * kXC * 2);  // xb then xcb
  float*          h1pre = (float*)alloc((size_t)kN * kHID * 4);          // then h2pre
  float*          h1    = (float*)alloc((size_t)kN * kHID * 4);
  float*          h2    = (float*)alloc((size_t)kN * kOUT * 4);
  float*          expe  = (float*)alloc((size_t)kEtot * kH * 4);
  float*          as_   = (float*)alloc((size_t)kN * kH * 4);
  float*          ad_   = (float*)alloc((size_t)kN * kH * 4);
  float*          emax  = (float*)alloc((size_t)kN * kH * 4);
  float*          denom = (float*)alloc((size_t)kN * kH * 4);
  unsigned short* W1t   = (unsigned short*)alloc((size_t)kIN * kHID * 2); // [HID][IN]
  unsigned short* W2t   = (unsigned short*)alloc((size_t)kXC * kOUT * 2); // [OUT][XC]
  int*            roots = (int*)alloc((size_t)kB * 4);
  float*          cnts  = (float*)alloc((size_t)kB * 4);

  const int T = 256;
  const float ninf = -__builtin_inff();

  // ---- convert inputs to bf16 (weights transposed for the WMMA B layout) ----
  k_cvt_bf16<<<cdiv(kN * kIN, T), T, 0, stream>>>(x, xb, kN * kIN);
  k_cvt_bf16_t<<<cdiv(kIN * kHID, T), T, 0, stream>>>(W1, W1t, kIN, kHID);
  k_cvt_bf16_t<<<cdiv(kXC * kOUT, T), T, 0, stream>>>(W2, W2t, kXC, kOUT);
  (void)hipMemsetAsync(h1, 0, (size_t)kN * kHID * 4, stream);
  (void)hipMemsetAsync(h2, 0, (size_t)kN * kOUT * 4, stream);

  // ---- layer 1: h1pre = x @ W1 (bf16 WMMA, fp32 accum) ----
  k_gemm_bf16<<<dim3(cdiv(kN, GBM), kHID / GBN), T, 0, stream>>>(xb, W1t, h1pre, kN, kHID, kIN);
  k_att<<<cdiv(kN * kH, T), T, 0, stream>>>(h1pre, att_s1, att_d1, as_, ad_, kD1);

  k_init_f32<<<cdiv(kN * kH, T), T, 0, stream>>>(emax, ninf, kN * kH);
  (void)hipMemsetAsync(denom, 0, (size_t)kN * kH * 4, stream);
  k_edge_max<<<cdiv(kEtot * kH, T), T, 0, stream>>>(ei, as_, ad_, emax);
  k_edge_exp<<<cdiv(kEtot * kH, T), T, 0, stream>>>(ei, as_, ad_, emax, denom, expe);
  k_edge_agg<<<cdiv(kEtot, 8), T, 0, stream>>>(ei, h1pre, expe, denom, h1, kHID, kD1);
  k_bias<<<cdiv(kN * kHID, T), T, 0, stream>>>(h1, b1, kHID, 0);

  // ---- roots + xc = relu(concat(h1, x[root])) in bf16 (reuse xb region) ----
  k_init_i32<<<cdiv(kB, T), T, 0, stream>>>(roots, 0x7fffffff, kB);
  k_roots<<<cdiv(kN, T), T, 0, stream>>>(batch, roots);
  k_build_xc<<<cdiv(kN * kXC, T), T, 0, stream>>>(h1, x, batch, roots, xb);

  // ---- layer 2: h2pre = xc @ W2 (reuse h1pre region) ----
  k_gemm_bf16<<<dim3(cdiv(kN, GBM), kOUT / GBN), T, 0, stream>>>(xb, W2t, h1pre, kN, kOUT, kXC);
  k_att<<<cdiv(kN * kH, T), T, 0, stream>>>(h1pre, att_s2, att_d2, as_, ad_, kD2);

  k_init_f32<<<cdiv(kN * kH, T), T, 0, stream>>>(emax, ninf, kN * kH);
  (void)hipMemsetAsync(denom, 0, (size_t)kN * kH * 4, stream);
  k_edge_max<<<cdiv(kEtot * kH, T), T, 0, stream>>>(ei, as_, ad_, emax);
  k_edge_exp<<<cdiv(kEtot * kH, T), T, 0, stream>>>(ei, as_, ad_, emax, denom, expe);
  k_edge_agg<<<cdiv(kEtot, 8), T, 0, stream>>>(ei, h1pre, expe, denom, h2, kOUT, kD2);
  k_bias<<<cdiv(kN * kOUT, T), T, 0, stream>>>(h2, b2, kOUT, 1);

  // ---- mean pool of concat(h2, h1[root]) per graph ----
  (void)hipMemsetAsync(out, 0, (size_t)kB * kFEAT * 4, stream);
  (void)hipMemsetAsync(cnts, 0, (size_t)kB * 4, stream);
  k_pool<<<cdiv(kN * kFEAT, T), T, 0, stream>>>(h2, h1, batch, roots, out, cnts);
  k_pool_div<<<cdiv(kB * kFEAT, T), T, 0, stream>>>(out, cnts);
}
